// UniversalGraphConv_10934986735712
// MI455X (gfx1250) — matrix operand order repaired and checked by reading, not compiled, not run on hardware
//
#include <hip/hip_runtime.h>
#include <hip/hip_bf16.h>

// ---------------------------------------------------------------------------
// Graph transformer layer for MI455X (gfx1250, wave32, WMMA, async-LDS).
//   qkv = X @ Wqkv^T + b ; masked 4-head attention (flash-style, bit mask,
//   4 waves share K/V staged in LDS via GLOBAL_LOAD_ASYNC_TO_LDS_B128);
//   att = ctx @ Wo^T + bo ; out = att @ Wp^T + bp + X ; LayerNorm.
// ---------------------------------------------------------------------------

#define NN 4096
#define DD 256
#define HH 4
#define HD 64
#define EE 131072
#define MWPR (NN / 32)          // mask words per row = 128
#define VPAD 40                 // padded V^T row length (80B, 16B aligned)

typedef __attribute__((ext_vector_type(16))) _Float16 v16h;
typedef __attribute__((ext_vector_type(8)))  float    v8f;

union Frag { v16h v; float4 q[2]; };

// A-matrix fragment (16 rows x 32 k), element (m,k) at base[m*stride + k].
__device__ __forceinline__ v16h loadA(const _Float16* base, int stride) {
  int lane = threadIdx.x & 31;
  const _Float16* p = base + (lane & 15) * stride + ((lane >> 4) << 3);
  Frag u;
  u.q[0] = *(const float4*)(p);
  u.q[1] = *(const float4*)(p + 16);
  return u.v;
}

// B-matrix fragment (32 k x 16 cols), element (k,n) at base[n*stride + k].
__device__ __forceinline__ v16h loadB(const _Float16* base, int stride) {
  int lane = threadIdx.x & 31;
  const _Float16* p = base + (lane & 15) * stride + ((lane >> 4) << 3);
  Frag u;
  u.q[0] = *(const float4*)(p);
  u.q[1] = *(const float4*)(p + 16);
  return u.v;
}

__device__ __forceinline__ v8f wmma16(v16h a, v16h b, v8f c) {
  return __builtin_amdgcn_wmma_f32_16x16x32_f16(false, a, false, b,
                                                (short)0, c, false, false);
}

// ---- CDNA5 async global->LDS copy (ASYNCcnt path, bypasses VGPRs) --------
__device__ __forceinline__ unsigned ldsAddr(const void* p) {
  return (unsigned)(unsigned long long)(const __attribute__((address_space(3))) void*)p;
}
__device__ __forceinline__ void asyncCopyB128(unsigned lds_byte, const void* g) {
  asm volatile("global_load_async_to_lds_b128 %0, %1, off"
               :: "v"(lds_byte), "v"(g) : "memory");
}
__device__ __forceinline__ void waitAsync() {
  asm volatile("s_wait_asynccnt 0x0" ::: "memory");
}

// ---------------------------------------------------------------- utilities
__global__ void k_f32_to_f16(const float* __restrict__ in,
                             _Float16* __restrict__ out, int n) {
  int i = blockIdx.x * blockDim.x + threadIdx.x;
  if (i < n) out[i] = (_Float16)in[i];
}

__global__ void k_mask_zero(unsigned int* __restrict__ mask, int words) {
  int i = blockIdx.x * blockDim.x + threadIdx.x;
  if (i < words) mask[i] = 0u;
}

__global__ void k_mask_edges(const long long* __restrict__ ei,
                             unsigned int* __restrict__ mask) {
  int e = blockIdx.x * blockDim.x + threadIdx.x;
  if (e < EE) {
    int s = (int)ei[e];
    int d = (int)ei[EE + e];
    atomicOr(&mask[s * MWPR + (d >> 5)], 1u << (d & 31));
    atomicOr(&mask[d * MWPR + (s >> 5)], 1u << (s & 31));
  }
}

__global__ void k_mask_diag(unsigned int* __restrict__ mask) {
  int i = blockIdx.x * blockDim.x + threadIdx.x;
  if (i < NN) atomicOr(&mask[i * MWPR + (i >> 5)], 1u << (i & 31));
}

// ------------------------------------------------------------- WMMA GEMM
// out[M,Nout](f16) = A[M,K](f16) @ W[Nout,K](f16)^T + bias.  One wave per
// 16x64 tile; all fragments straight from global (layouts are contiguous).
__global__ __launch_bounds__(32) void k_gemm(
    const _Float16* __restrict__ A, const _Float16* __restrict__ W,
    const float* __restrict__ bias, _Float16* __restrict__ out,
    int M, int K, int Nout) {
  int row0 = blockIdx.x * 16;
  int lane = threadIdx.x;
  int n = lane & 15, hh = lane >> 4;
  v8f acc[4] = {};
  for (int kb = 0; kb < K; kb += 32) {
    v16h a = loadA(A + row0 * K + kb, K);
#pragma unroll
    for (int j = 0; j < 4; ++j) {
      int col0 = blockIdx.y * 64 + j * 16;
      v16h b = loadB(W + col0 * K + kb, K);
      acc[j] = wmma16(a, b, acc[j]);
    }
  }
#pragma unroll
  for (int j = 0; j < 4; ++j) {
    int col = blockIdx.y * 64 + j * 16 + n;
    float bv = bias[col];
#pragma unroll
    for (int r = 0; r < 8; ++r) {
      int row = row0 + r + 8 * hh;
      out[row * Nout + col] = (_Float16)(acc[j][r] + bv);
    }
  }
}

// ------------------------------------------------------- flash attention
// 4 waves / workgroup, 64 queries per block, one head per blockIdx.y.
// K blocks async-copied global->LDS (double buffered); V blocks staged
// transposed cooperatively so P.V B-fragments are contiguous ds_load_b128.
__global__ __launch_bounds__(128) void k_attn(
    const _Float16* __restrict__ qkv,        // [N, 768] = Q|K|V per row
    const unsigned int* __restrict__ mask,   // bit-packed N x N
    _Float16* __restrict__ ctx) {            // [N, 256] head-interleaved
  int h    = blockIdx.y;
  int tid  = threadIdx.x;
  int wave = tid >> 5;
  int lane = tid & 31;
  int n = lane & 15, hh = lane >> 4;
  int q0 = blockIdx.x * 64 + wave * 16;

  __shared__ __align__(16) _Float16 kbuf[2][32][HD];    // 8 KB  (row-major K)
  __shared__ __align__(16) _Float16 vt[2][HD][VPAD];    // 10 KB (V^T, padded)
  __shared__ __align__(16) _Float16 pbuf[4][16][32];    // 4 KB  (per-wave P)

  // ---- staging helpers -------------------------------------------------
  auto stageK = [&](int buf, int kb) {
    // 32 keys x 128B = 256 x 16B chunks over 128 threads (2 each), async.
#pragma unroll
    for (int c = 0; c < 2; ++c) {
      int chunk = tid * 2 + c;                 // 0..255
      int key = chunk >> 3, seg = (chunk & 7) << 3;   // seg in halves
      const _Float16* g = qkv + (size_t)(kb + key) * 768 + 256 + h * HD + seg;
      asyncCopyB128(ldsAddr(&kbuf[buf][key][seg]), g);
    }
  };
  auto stageV = [&](int buf, int kb) {
    // register transpose: thread owns (key, 16-dim segment)
    int key = tid & 31, dseg = (tid >> 5) << 4;       // 0,16,32,48
    const _Float16* g = qkv + (size_t)(kb + key) * 768 + 512 + h * HD + dseg;
    float4 t0 = *(const float4*)(g);
    float4 t1 = *(const float4*)(g + 8);
    const _Float16* h0 = (const _Float16*)&t0;
    const _Float16* h1 = (const _Float16*)&t1;
#pragma unroll
    for (int j = 0; j < 8; ++j) vt[buf][dseg + j][key]     = h0[j];
#pragma unroll
    for (int j = 0; j < 8; ++j) vt[buf][dseg + 8 + j][key] = h1[j];
  };

  // ---- per-wave Q fragments (d = 0..31 / 32..63) ------------------------
  v16h qa0 = loadA(qkv + (size_t)q0 * 768 + h * HD, 768);
  v16h qa1 = loadA(qkv + (size_t)q0 * 768 + h * HD + 32, 768);

  float m[8], l[8];
  v8f acc[4] = {};
#pragma unroll
  for (int r = 0; r < 8; ++r) { m[r] = -1e30f; l[r] = 0.0f; }

  stageK(0, 0);
  stageV(0, 0);
  waitAsync();
  __syncthreads();

  for (int kb = 0; kb < NN; kb += 32) {
    int cur = (kb >> 5) & 1, nxt = cur ^ 1;
    if (kb + 32 < NN) { stageK(nxt, kb + 32); stageV(nxt, kb + 32); }

    // ---- scores: S[16x32] = Q . K^T from LDS ----
    v8f s[2] = {};
#pragma unroll
    for (int kt = 0; kt < 2; ++kt) {
      v16h b0 = loadB(&kbuf[cur][kt * 16][0], HD);
      v16h b1 = loadB(&kbuf[cur][kt * 16][32], HD);
      s[kt] = wmma16(qa0, b0, s[kt]);
      s[kt] = wmma16(qa1, b1, s[kt]);
    }

    // one mask word covers this 32-key block for query row n
    int mw = (int)mask[(size_t)(q0 + n) * MWPR + (kb >> 5)];

    // ---- online softmax, write P(f16) to this wave's pbuf ----
#pragma unroll
    for (int r = 0; r < 8; ++r) {
      int qrow = r + 8 * hh;
      unsigned int w = (unsigned int)__shfl(mw, qrow, 16);
      float s0 = s[0][r] * 0.125f;                 // 1/sqrt(64)
      float s1 = s[1][r] * 0.125f;
      if (!((w >> n) & 1u))        s0 = -1e30f;
      if (!((w >> (16 + n)) & 1u)) s1 = -1e30f;
      float mx = fmaxf(s0, s1);
#pragma unroll
      for (int off = 1; off < 16; off <<= 1)
        mx = fmaxf(mx, __shfl_xor(mx, off, 16));
      float mnew = fmaxf(m[r], mx);
      float alpha = __expf(m[r] - mnew);
      float p0 = (s0 <= -1e29f) ? 0.0f : __expf(s0 - mnew);
      float p1 = (s1 <= -1e29f) ? 0.0f : __expf(s1 - mnew);
      float rs = p0 + p1;
#pragma unroll
      for (int off = 1; off < 16; off <<= 1)
        rs += __shfl_xor(rs, off, 16);
      l[r] = l[r] * alpha + rs;
      m[r] = mnew;
#pragma unroll
      for (int dt = 0; dt < 4; ++dt) acc[dt][r] *= alpha;
      pbuf[wave][qrow][n]      = (_Float16)p0;
      pbuf[wave][qrow][16 + n] = (_Float16)p1;
    }

    // ---- ctx[16x64] += P[16x32] . V[32x64] (wave-local pbuf, shared vt) --
    v16h pa = loadA(&pbuf[wave][0][0], 32);
#pragma unroll
    for (int dt = 0; dt < 4; ++dt) {
      v16h vb = loadB(&vt[cur][dt * 16][0], VPAD);
      acc[dt] = wmma16(pa, vb, acc[dt]);
    }

    waitAsync();        // our async K copies into nxt have landed
    __syncthreads();    // all staging visible; all reads of cur done
  }

  // ---- normalize and store ----
#pragma unroll
  for (int r = 0; r < 8; ++r) {
    float inv = 1.0f / l[r];
    int row = q0 + r + 8 * hh;
#pragma unroll
    for (int dt = 0; dt < 4; ++dt)
      ctx[(size_t)row * DD + h * HD + dt * 16 + n] = (_Float16)(acc[dt][r] * inv);
  }
}

// ---------------------------------- fused proj + bias + residual + layernorm
__global__ __launch_bounds__(32) void k_proj_ln(
    const _Float16* __restrict__ att,   // [N,256] f16
    const _Float16* __restrict__ Wp,    // [256,256] f16
    const float* __restrict__ bp,
    const float* __restrict__ x,        // residual [N,256] f32
    const float* __restrict__ gamma, const float* __restrict__ beta,
    float* __restrict__ out) {
  int m0 = blockIdx.x * 16;
  int lane = threadIdx.x;
  int n = lane & 15, hh = lane >> 4;
  __shared__ float y[16][DD];

  v16h afr[8];
#pragma unroll
  for (int kb = 0; kb < 8; ++kb)
    afr[kb] = loadA(att + (size_t)m0 * DD + kb * 32, DD);

  for (int nt = 0; nt < 16; ++nt) {
    v8f acc = {};
#pragma unroll
    for (int kb = 0; kb < 8; ++kb) {
      v16h b = loadB(Wp + (size_t)(nt * 16) * DD + kb * 32, DD);
      acc = wmma16(afr[kb], b, acc);
    }
    int col = nt * 16 + n;
    float bv = bp[col];
#pragma unroll
    for (int r = 0; r < 8; ++r) {
      int row = r + 8 * hh;
      y[row][col] = acc[r] + bv + x[(size_t)(m0 + row) * DD + col];
    }
  }
  __syncthreads();

  for (int row = 0; row < 16; ++row) {
    float s = 0.0f, s2 = 0.0f;
    for (int c = lane; c < DD; c += 32) {
      float v = y[row][c];
      s += v; s2 += v * v;
    }
#pragma unroll
    for (int off = 1; off < 32; off <<= 1) {
      s  += __shfl_xor(s,  off, 32);
      s2 += __shfl_xor(s2, off, 32);
    }
    float mu  = s * (1.0f / DD);
    float var = s2 * (1.0f / DD) - mu * mu;
    float rinv = rsqrtf(var + 1e-5f);
    for (int c = lane; c < DD; c += 32)
      out[(size_t)(m0 + row) * DD + c] = (y[row][c] - mu) * rinv * gamma[c] + beta[c];
  }
}

// ---------------------------------------------------------------------------
extern "C" void kernel_launch(void* const* d_in, const int* in_sizes, int n_in,
                              void* d_out, int out_size, void* d_ws,
                              size_t ws_size, hipStream_t stream) {
  const float*     x      = (const float*)d_in[0];       // [N,256]
  const long long* eidx   = (const long long*)d_in[1];   // [2,E] int64
  const float*     w_qkv  = (const float*)d_in[2];       // [768,256]
  const float*     b_qkv  = (const float*)d_in[3];       // [768]
  const float*     w_out  = (const float*)d_in[4];       // [256,256]
  const float*     b_out  = (const float*)d_in[5];       // [256]
  const float*     w_proj = (const float*)d_in[6];       // [256,256]
  const float*     b_proj = (const float*)d_in[7];       // [256]
  const float*     gamma  = (const float*)d_in[8];
  const float*     beta   = (const float*)d_in[9];
  float* out = (float*)d_out;

  size_t off = 0;
  auto carve = [&](size_t bytes) {
    void* p = (char*)d_ws + off;
    off += (bytes + 255) & ~(size_t)255;
    return p;
  };
  unsigned int* mask = (unsigned int*)carve((size_t)NN * MWPR * 4);  // 2 MB
  _Float16* Xh   = (_Float16*)carve((size_t)NN * DD * 2);            // 2 MB
  _Float16* Wqkv = (_Float16*)carve((size_t)3 * DD * DD * 2);
  _Float16* Wo   = (_Float16*)carve((size_t)DD * DD * 2);
  _Float16* Wp   = (_Float16*)carve((size_t)DD * DD * 2);
  _Float16* QKV  = (_Float16*)carve((size_t)NN * 3 * DD * 2);        // 6 MB
  _Float16* CTX  = (_Float16*)carve((size_t)NN * DD * 2);
  _Float16* ATT  = (_Float16*)carve((size_t)NN * DD * 2);

  // 1) fp32 -> fp16 conversions
  k_f32_to_f16<<<(NN * DD + 255) / 256, 256, 0, stream>>>(x, Xh, NN * DD);
  k_f32_to_f16<<<(3 * DD * DD + 255) / 256, 256, 0, stream>>>(w_qkv, Wqkv, 3 * DD * DD);
  k_f32_to_f16<<<(DD * DD + 255) / 256, 256, 0, stream>>>(w_out, Wo, DD * DD);
  k_f32_to_f16<<<(DD * DD + 255) / 256, 256, 0, stream>>>(w_proj, Wp, DD * DD);

  // 2) bit-packed allowed mask (edges symmetrized + self loops)
  int words = NN * MWPR;
  k_mask_zero<<<(words + 255) / 256, 256, 0, stream>>>(mask, words);
  k_mask_edges<<<(EE + 255) / 256, 256, 0, stream>>>(eidx, mask);
  k_mask_diag<<<(NN + 255) / 256, 256, 0, stream>>>(mask);

  // 3) QKV projection: [N,768] = Xh @ Wqkv^T + b
  k_gemm<<<dim3(NN / 16, (3 * DD) / 64), 32, 0, stream>>>(
      Xh, Wqkv, b_qkv, QKV, NN, DD, 3 * DD);

  // 4) masked flash attention -> CTX [N,256]
  k_attn<<<dim3(NN / 64, HH), 128, 0, stream>>>(QKV, mask, CTX);

  // 5) output projection: ATT = CTX @ Wo^T + bo
  k_gemm<<<dim3(NN / 16, DD / 64), 32, 0, stream>>>(
      CTX, Wo, b_out, ATT, NN, DD, DD);

  // 6) proj + residual + layernorm -> out (f32)
  k_proj_ln<<<NN / 16, 32, 0, stream>>>(ATT, Wp, b_proj, x, gamma, beta, out);
}
